// RNN_LSTM_824633721244
// MI455X (gfx1250) — matrix operand order repaired and checked by reading, not compiled, not run on hardware
//
#include <hip/hip_runtime.h>
#include <hip/hip_bf16.h>

typedef __attribute__((ext_vector_type(16))) __bf16 v16bf;
typedef __attribute__((ext_vector_type(8)))  float  v8f;

#define F_IN   22000
#define K_MAIN 21984    // F_IN & ~31 : last full-free slab boundary
#define D_N    100
#define D_NP   112      // N padded to 7 tiles of 16
#define HID    128
#define G4     512      // 4*H
#define BATCH  4096
#define TSTEP  100
#define MB     32       // batch rows per LSTM workgroup

#if __has_builtin(__builtin_amdgcn_tanhf)
#define TANHF(x) __builtin_amdgcn_tanhf(x)
#elif __has_builtin(__builtin_amdgcn_tanh_f32)
#define TANHF(x) __builtin_amdgcn_tanh_f32(x)
#else
#define TANHF(x) tanhf(x)
#endif

__device__ __forceinline__ float sigmoid_(float x) {
    return 0.5f + 0.5f * TANHF(0.5f * x);   // v_tanh_f32 path on gfx1250
}

// ---------------------------------------------------------------------------
// One k-slab (32 wide) of the Dense(100) GEMM. GUARD=true only for the tail.
// Branch-free staging: clamp index + 0/1 scale instead of exec-mask branches.
// ---------------------------------------------------------------------------
template <bool GUARD>
__device__ __forceinline__ void dense_slab(
    const int k0, const int tid, const int ln, const int lh,
    const float* __restrict__ xrow, const float* __restrict__ W1,
    __bf16* __restrict__ W1s, v8f* __restrict__ acc)
{
    __syncthreads();                        // previous slab fully consumed
    #pragma unroll
    for (int i = 0; i < 7; ++i) {           // 1792 pairs / 256 threads
        int p   = tid + i * 256;
        int n   = p % D_NP;
        int kk2 = (p / D_NP) * 2;
        int   ncl = (n < D_N) ? n : 0;
        float sn  = (n < D_N) ? 1.0f : 0.0f;
        float v0, v1;
        if (GUARD) {
            int k = k0 + kk2;
            int   kc0 = (k     < F_IN) ? k     : 0;
            int   kc1 = (k + 1 < F_IN) ? k + 1 : 0;
            float s0  = (k     < F_IN) ? sn : 0.0f;
            float s1  = (k + 1 < F_IN) ? sn : 0.0f;
            v0 = W1[(size_t)kc0 * D_N + ncl] * s0;
            v1 = W1[(size_t)kc1 * D_N + ncl] * s1;
        } else {
            v0 = W1[(size_t)(k0 + kk2)     * D_N + ncl] * sn;
            v1 = W1[(size_t)(k0 + kk2 + 1) * D_N + ncl] * sn;
        }
        W1s[n * 32 + kk2]     = (__bf16)v0;
        W1s[n * 32 + kk2 + 1] = (__bf16)v1;
    }
    __syncthreads();

    __builtin_prefetch(xrow + k0 + 128, 0, 1);   // global_prefetch_b8

    // A fragment: two aligned 8-float runs (only tail needs a guard)
    v16bf a;
    #pragma unroll
    for (int half = 0; half < 2; ++half) {
        int kb = k0 + lh * 16 + half * 8;
        float4 p0, p1;
        if (!GUARD || kb < F_IN) {
            p0 = *reinterpret_cast<const float4*>(xrow + kb);
            p1 = *reinterpret_cast<const float4*>(xrow + kb + 4);
        } else {
            p0 = float4{0.f, 0.f, 0.f, 0.f};
            p1 = p0;
        }
        int e = half * 8;
        a[e + 0] = (__bf16)p0.x;  a[e + 1] = (__bf16)p0.y;
        a[e + 2] = (__bf16)p0.z;  a[e + 3] = (__bf16)p0.w;
        a[e + 4] = (__bf16)p1.x;  a[e + 5] = (__bf16)p1.y;
        a[e + 6] = (__bf16)p1.z;  a[e + 7] = (__bf16)p1.w;
    }

    // Load ALL 7 B fragments first (14 ds_load_b128 in flight), then 7 WMMAs
    v16bf bfr[7];
    #pragma unroll
    for (int nt = 0; nt < 7; ++nt) {
        const __bf16* col = &W1s[(nt * 16 + ln) * 32 + lh * 16];
        #pragma unroll
        for (int e = 0; e < 16; ++e) bfr[nt][e] = col[e];
    }
    #pragma unroll
    for (int nt = 0; nt < 7; ++nt)
        acc[nt] = __builtin_amdgcn_wmma_f32_16x16x32_bf16(
            false, a, false, bfr[nt], (short)0, acc[nt], false, false);
}

// ---------------------------------------------------------------------------
// Kernel 1: d[4096,100] = x[4096,22000] @ W1[22000,100] + b1
// ---------------------------------------------------------------------------
__global__ __launch_bounds__(256) void dense1_kernel(
    const float* __restrict__ x, const float* __restrict__ W1,
    const float* __restrict__ b1, float* __restrict__ d)
{
    __shared__ __bf16 W1s[D_NP * 32];         // [n][kk] bf16, 7 KB

    const int tid  = threadIdx.x;
    const int wave = tid >> 5;
    const int lane = tid & 31;
    const int m0   = (blockIdx.x * 8 + wave) * 16;
    const int ln   = lane & 15;
    const int lh   = lane >> 4;

    v8f acc[7];
    #pragma unroll
    for (int nt = 0; nt < 7; ++nt)
        #pragma unroll
        for (int e = 0; e < 8; ++e) acc[nt][e] = 0.0f;

    const float* xrow = x + (size_t)(m0 + ln) * F_IN;

    for (int k0 = 0; k0 < K_MAIN; k0 += 32)                    // guard-free
        dense_slab<false>(k0, tid, ln, lh, xrow, W1, W1s, acc);
    dense_slab<true>(K_MAIN, tid, ln, lh, xrow, W1, W1s, acc); // tail

    #pragma unroll
    for (int nt = 0; nt < 7; ++nt) {
        #pragma unroll
        for (int r = 0; r < 8; ++r) {
            int m = m0 + lh * 8 + r;
            int n = nt * 16 + ln;
            if (n < D_N) d[(size_t)m * D_N + n] = acc[nt][r] + b1[n];
        }
    }
}

// ---------------------------------------------------------------------------
// Kernel 2: LSTM recurrence. 128 WGs x 32 batch rows, 8 waves/WG.
// Wr cached whole in LDS TRANSPOSED as bf16 [g][k] (128 KB); h ping-pongs
// through LDS (bf16); cell state in VGPRs; Wk/bl hoisted to registers.
// ---------------------------------------------------------------------------
__global__ __launch_bounds__(256) void lstm_kernel(
    const float* __restrict__ dvec,   // [4096,100] dense output
    const float* __restrict__ Wk,     // [1,512]
    const float* __restrict__ Wr,     // [128,512]
    const float* __restrict__ bl,     // [512]
    float* __restrict__ hout)         // [4096,128]
{
    extern __shared__ char smem[];
    __bf16* WrT_s = (__bf16*)smem;                 // [512][128] bf16 = 128 KB
    __bf16* h_s   = (__bf16*)(smem + 131072);      // [32][128] bf16 = 8 KB
    float*  dcol  = (float*)(smem + 131072 + 8192); // 32 f32 (d[:,t] slab)

    const int tid  = threadIdx.x;
    const int wave = tid >> 5;          // 0..7 == unit tile
    const int lane = tid & 31;
    const int ln   = lane & 15;
    const int lh   = lane >> 4;
    const int b0   = blockIdx.x * MB;

    for (int idx = tid; idx < HID * G4; idx += 256) {
        int k = idx >> 9;               // Wr row
        int g = idx & 511;              // Wr col
        WrT_s[g * HID + k] = (__bf16)Wr[idx];
    }
    for (int idx = tid; idx < MB * HID; idx += 256) h_s[idx] = (__bf16)0.0f;
    __syncthreads();

    const int u = wave * 16 + ln;       // unit column this lane produces
    // per-lane constants: input-kernel and bias for the 4 gates of unit u
    const float wk0 = Wk[u],           wk1 = Wk[HID + u];
    const float wk2 = Wk[2 * HID + u], wk3 = Wk[3 * HID + u];
    const float bl0 = bl[u],           bl1 = bl[HID + u];
    const float bl2 = bl[2 * HID + u], bl3 = bl[3 * HID + u];

    v8f cstate[2];
    #pragma unroll
    for (int mt = 0; mt < 2; ++mt)
        #pragma unroll
        for (int e = 0; e < 8; ++e) cstate[mt][e] = 0.0f;

    for (int t = 0; t < TSTEP; ++t) {
        if (tid < MB) dcol[tid] = dvec[(size_t)(b0 + tid) * D_N + t];

        v8f acc[2][4];
        #pragma unroll
        for (int mt = 0; mt < 2; ++mt)
            #pragma unroll
            for (int q = 0; q < 4; ++q)
                #pragma unroll
                for (int e = 0; e < 8; ++e) acc[mt][q][e] = 0.0f;

        #pragma unroll
        for (int ks = 0; ks < 4; ++ks) {
            const int k0 = ks * 32;

            v16bf a[2];
            #pragma unroll
            for (int mt = 0; mt < 2; ++mt) {
                const __bf16* hr = &h_s[(mt * 16 + ln) * HID];
                #pragma unroll
                for (int half = 0; half < 2; ++half) {
                    const __bf16* run = hr + k0 + lh * 16 + half * 8;
                    #pragma unroll
                    for (int e = 0; e < 8; ++e) a[mt][half * 8 + e] = run[e];
                }
            }

            v16bf bg[4];
            #pragma unroll
            for (int q = 0; q < 4; ++q) {
                const int g = q * HID + wave * 16 + ln;
                const __bf16* run = &WrT_s[g * HID + k0 + lh * 16];
                #pragma unroll
                for (int e = 0; e < 16; ++e) bg[q][e] = run[e];
            }

            #pragma unroll
            for (int mt = 0; mt < 2; ++mt)
                #pragma unroll
                for (int q = 0; q < 4; ++q)
                    acc[mt][q] = __builtin_amdgcn_wmma_f32_16x16x32_bf16(
                        false, a[mt], false, bg[q], (short)0, acc[mt][q],
                        false, false);
        }
        __syncthreads();   // all h_s reads done before overwrite; dcol visible

        #pragma unroll
        for (int mt = 0; mt < 2; ++mt) {
            const int mloc = mt * 16 + lh * 8;
            #pragma unroll
            for (int r = 0; r < 8; ++r) {
                const int b  = b0 + mloc + r;
                const float dv = dcol[mloc + r];
                float iv = sigmoid_(acc[mt][0][r] + dv * wk0 + bl0);
                float fv = sigmoid_(acc[mt][1][r] + dv * wk1 + bl1);
                float gv = TANHF  (acc[mt][2][r] + dv * wk2 + bl2);
                float ov = sigmoid_(acc[mt][3][r] + dv * wk3 + bl3);
                float c  = fv * cstate[mt][r] + iv * gv;
                cstate[mt][r] = c;
                float h = ov * TANHF(c);
                h_s[(mloc + r) * HID + u] = (__bf16)h;
                if (t == TSTEP - 1) hout[(size_t)b * HID + u] = h;
            }
        }
        __syncthreads();   // h_s fully updated before next step's reads
    }
}

// ---------------------------------------------------------------------------
// Kernel 3: logits = h @ W2 + b2, softmax. One thread per batch row.
// ---------------------------------------------------------------------------
__global__ __launch_bounds__(256) void head_kernel(
    const float* __restrict__ h, const float* __restrict__ W2,
    const float* __restrict__ b2, float* __restrict__ out)
{
    int b = blockIdx.x * blockDim.x + threadIdx.x;
    if (b >= BATCH) return;
    float acc[4] = {b2[0], b2[1], b2[2], b2[3]};
    const float* hr = h + (size_t)b * HID;
    for (int uu = 0; uu < HID; uu += 4) {
        float4 hv = *reinterpret_cast<const float4*>(hr + uu);
        #pragma unroll
        for (int c = 0; c < 4; ++c) {
            acc[c] += hv.x * W2[(uu + 0) * 4 + c];
            acc[c] += hv.y * W2[(uu + 1) * 4 + c];
            acc[c] += hv.z * W2[(uu + 2) * 4 + c];
            acc[c] += hv.w * W2[(uu + 3) * 4 + c];
        }
    }
    float mx = fmaxf(fmaxf(acc[0], acc[1]), fmaxf(acc[2], acc[3]));
    float s = 0.0f;
    #pragma unroll
    for (int c = 0; c < 4; ++c) { acc[c] = __expf(acc[c] - mx); s += acc[c]; }
    float inv = 1.0f / s;
    #pragma unroll
    for (int c = 0; c < 4; ++c) out[(size_t)b * 4 + c] = acc[c] * inv;
}

extern "C" void kernel_launch(void* const* d_in, const int* in_sizes, int n_in,
                              void* d_out, int out_size, void* d_ws, size_t ws_size,
                              hipStream_t stream) {
    const float* x  = (const float*)d_in[0];
    const float* W1 = (const float*)d_in[1];
    const float* b1 = (const float*)d_in[2];
    const float* Wk = (const float*)d_in[3];
    const float* Wr = (const float*)d_in[4];
    const float* bl = (const float*)d_in[5];
    const float* W2 = (const float*)d_in[6];
    const float* b2 = (const float*)d_in[7];
    float* out = (float*)d_out;

    float* d_dense = (float*)d_ws;                                        // 4096*100 f32
    float* d_hN    = (float*)((char*)d_ws + (size_t)BATCH * D_N * 4);     // 4096*128 f32

    dense1_kernel<<<dim3(BATCH / 128), dim3(256), 0, stream>>>(x, W1, b1, d_dense);

    const int lstm_smem = 131072 + 8192 + 128;
    hipFuncSetAttribute(reinterpret_cast<const void*>(lstm_kernel),
                        hipFuncAttributeMaxDynamicSharedMemorySize, lstm_smem);
    lstm_kernel<<<dim3(BATCH / MB), dim3(256), lstm_smem, stream>>>(
        d_dense, Wk, Wr, bl, d_hN);

    head_kernel<<<dim3(BATCH / 256), dim3(256), 0, stream>>>(d_hN, W2, b2, out);
}
